// AttGRU_13417477833016
// MI455X (gfx1250) — matrix operand order repaired and checked by reading, not compiled, not run on hardware
//
#include <hip/hip_runtime.h>
#include <hip/hip_bf16.h>
#include <math.h>

#define Hq 100
#define Sq 1024
#define Bq 512
#define PITCH 400   // halves per A-tile row: 128 (c) + 128 (h buf0) + 128 (h buf1) + 16 pad

typedef __attribute__((ext_vector_type(16))) _Float16 v16h;
typedef __attribute__((ext_vector_type(8)))  _Float16 v8h;
typedef __attribute__((ext_vector_type(8)))  float    v8f;
typedef __attribute__((ext_vector_type(4)))  float    v4f;

// fast device transcendentals: single v_rcp_f32 / v_exp_f32 / v_tanh_f32
__device__ __forceinline__ float fast_sigmoid(float x) {
  return __builtin_amdgcn_rcpf(1.f + __expf(-x));
}
__device__ __forceinline__ float fast_tanh(float x) {
#if __has_builtin(__builtin_amdgcn_tanhf)
  return __builtin_amdgcn_tanhf(x);   // gfx1250 v_tanh_f32
#else
  return 2.f * __builtin_amdgcn_rcpf(1.f + __expf(-2.f * x)) - 1.f;
#endif
}

// ---------------------------------------------------------------------------
// Kernel 1: attn = q.c per (b,s), softmax over s -> scores (written to d_out tail)
// ---------------------------------------------------------------------------
__global__ __launch_bounds__(256) void attn_softmax_kernel(
    const float* __restrict__ query,
    const float* __restrict__ context,
    float* __restrict__ scores)
{
  const int b   = blockIdx.x;
  const int tid = threadIdx.x;
  __shared__ float qs[Hq];
  __shared__ float red[256];
  if (tid < Hq) qs[tid] = query[b * Hq + tid];
  __syncthreads();

  float vals[4];
  float vmax = -1e30f;
  #pragma unroll
  for (int i = 0; i < 4; ++i) {
    const int s = tid + i * 256;
    const float* cp = context + ((size_t)b * Sq + s) * Hq;
    float acc = 0.f;
    #pragma unroll 4
    for (int k = 0; k < Hq; ++k) acc += qs[k] * cp[k];
    vals[i] = acc;
    vmax = fmaxf(vmax, acc);
  }
  red[tid] = vmax;
  __syncthreads();
  for (int off = 128; off > 0; off >>= 1) {
    if (tid < off) red[tid] = fmaxf(red[tid], red[tid + off]);
    __syncthreads();
  }
  vmax = red[0];
  __syncthreads();
  float vsum = 0.f;
  #pragma unroll
  for (int i = 0; i < 4; ++i) { vals[i] = __expf(vals[i] - vmax); vsum += vals[i]; }
  red[tid] = vsum;
  __syncthreads();
  for (int off = 128; off > 0; off >>= 1) {
    if (tid < off) red[tid] += red[tid + off];
    __syncthreads();
  }
  const float inv = __builtin_amdgcn_rcpf(red[0]);
  #pragma unroll
  for (int i = 0; i < 4; ++i)
    scores[(size_t)b * Sq + tid + i * 256] = vals[i] * inv;
}

// ---------------------------------------------------------------------------
// Kernel 2: fused bidirectional attention-GRU scan.
// grid = 64 WGs: blocks 0..31 forward, 32..63 backward; each WG owns 16 batch
// rows. 7 waves (224 threads); wave w owns output columns [16w, 16w+16).
// Per step: 16 x v_wmma_f32_16x16x32_f16 per wave over A=[c|h] (K=256).
// ---------------------------------------------------------------------------
__global__ __launch_bounds__(224, 1) void attgru_scan_kernel(
    const float* __restrict__ context,
    const float* __restrict__ init_hidden,
    const float* __restrict__ scores,
    const float* __restrict__ Wr_f, const float* __restrict__ br_f,
    const float* __restrict__ Ur_f, const float* __restrict__ bur_f,
    const float* __restrict__ W_f,  const float* __restrict__ bw_f,
    const float* __restrict__ U_f,  const float* __restrict__ bu_f,
    const float* __restrict__ Wr_b, const float* __restrict__ br_b,
    const float* __restrict__ Ur_b, const float* __restrict__ bur_b,
    const float* __restrict__ W_b,  const float* __restrict__ bw_b,
    const float* __restrict__ U_b,  const float* __restrict__ bu_b,
    float* __restrict__ out)
{
  const int dir  = blockIdx.x >> 5;     // 0 = forward, 1 = backward
  const int wg   = blockIdx.x & 31;
  const int b0   = wg * 16;             // first batch row of this WG
  const int tid  = threadIdx.x;
  const int wave = tid >> 5;            // 0..6 -> N-tile
  const int lane = tid & 31;
  const bool hi  = lane >= 16;
  const int m16  = lane & 15;           // A-operand row
  const int n    = wave * 16 + m16;     // output column of this lane
  const bool nok = n < Hq;
  const int mb   = hi ? 8 : 0;          // C/D rows: mb..mb+7

  const float* Wr = dir ? Wr_b : Wr_f;  const float* br  = dir ? br_b  : br_f;
  const float* Ur = dir ? Ur_b : Ur_f;  const float* bur = dir ? bur_b : bur_f;
  const float* W  = dir ? W_b  : W_f;   const float* bw  = dir ? bw_b  : bw_f;
  const float* U  = dir ? U_b  : U_f;   const float* bu  = dir ? bu_b  : bu_f;
  const float* h0 = init_hidden + (size_t)dir * Bq * Hq;

  __shared__ __align__(16) _Float16 At[16][PITCH];
  __shared__ __align__(16) float gb[128 * 16];  // gate chunk, [t_local][m]

  // zero the whole A tile (zeros the k / n padding and both h buffers)
  {
    _Float16* Af = &At[0][0];
    for (int idx = tid; idx < 16 * PITCH; idx += 224) Af[idx] = (_Float16)0.f;
  }

  // --- preload B operands into VGPRs (constant across all 1024 steps) ------
  // B1 = [Wr^T ; Ur^T] (K=256, 8 chunks), B2 = W^T (4), B3 = U^T (4).
  // Layout: lanes 0-15 hold K = kchunk+0..15, lanes 16-31 hold K = kchunk+16..31.
  v16h B1[8], B2[4], B3[4];
  #pragma unroll
  for (int j = 0; j < 8; ++j) {
    const int kb = j * 32 + (hi ? 16 : 0);
    v16h v;
    #pragma unroll
    for (int i = 0; i < 16; ++i) {
      const int k = kb + i;
      float f = 0.f;
      if (j < 4) { if (nok && k < Hq) f = Wr[n * Hq + k]; }
      else       { const int kk = k - 128; if (nok && kk < Hq) f = Ur[n * Hq + kk]; }
      v[i] = (_Float16)f;
    }
    B1[j] = v;
  }
  #pragma unroll
  for (int j = 0; j < 4; ++j) {
    const int kb = j * 32 + (hi ? 16 : 0);
    v16h v2, v3;
    #pragma unroll
    for (int i = 0; i < 16; ++i) {
      const int k = kb + i;
      float f2 = 0.f, f3 = 0.f;
      if (nok && k < Hq) { f2 = W[n * Hq + k]; f3 = U[n * Hq + k]; }
      v2[i] = (_Float16)f2; v3[i] = (_Float16)f3;
    }
    B2[j] = v2; B3[j] = v3;
  }

  const float bias1 = nok ? (br[n] + bur[n]) : 0.f;  // sigmoid pre-act bias
  const float bias2 = nok ? bw[n] : 0.f;             // c@W^T bias
  const float bias3 = nok ? bu[n] : 0.f;             // h@U^T bias

  __syncthreads();  // zero-fill visible before writing h0

  // initial hidden state -> registers + h buffer 0 (cols 128..255)
  float hreg[8];
  #pragma unroll
  for (int r = 0; r < 8; ++r) {
    const int m = mb + r;
    const float hv = nok ? h0[(size_t)(b0 + m) * Hq + n] : 0.f;
    hreg[r] = hv;
    At[m][128 + n] = (_Float16)hv;
  }

  // --------------------------- sequential scan -----------------------------
  for (int t = 0; t < Sq; ++t) {
    const int s = dir ? (Sq - 1 - t) : t;

    // refill the gate chunk every 128 steps (aligned for both directions)
    if ((t & 127) == 0) {
      const int s0 = dir ? (Sq - 128 - t) : t;
      for (int idx = tid; idx < 16 * 128; idx += 224) {
        const int j = idx >> 4, m = idx & 15;     // gb[t_local][m]
        gb[idx] = scores[(size_t)(b0 + m) * Sq + s0 + j];
      }
    }
    // cooperative load of c(t): 16 rows x 100 f32 -> f16, zero-padded to 128
    for (int idx = tid; idx < 16 * 128; idx += 224) {
      const int m = idx >> 7, k = idx & 127;
      const float f =
          (k < Hq) ? context[((size_t)(b0 + m) * Sq + s) * Hq + k] : 0.f;
      At[m][k] = (_Float16)f;
    }
    // prefetch next step's context rows (global_prefetch_b8 path)
    if (t + 1 < Sq && tid < 16) {
      const int s2 = dir ? (Sq - 2 - t) : (t + 1);
      __builtin_prefetch(&context[((size_t)(b0 + tid) * Sq + s2) * Hq], 0, 1);
    }
    __syncthreads();  // A tile (c + current h buffer + gates) ready

    const int hoff   = 128 + 128 * (t & 1);        // h buffer to read
    const int hstore = 128 + 128 * ((t + 1) & 1);  // h buffer to write

    v8f a1, a2, a3;
    #pragma unroll
    for (int r = 0; r < 8; ++r) { a1[r] = bias1; a2[r] = bias2; a3[r] = bias3; }

    #pragma unroll
    for (int j = 0; j < 8; ++j) {
      // A layout: lanes 0-15 -> K kc+0..15; lanes 16-31 -> K kc+16..31
      const int kb = (j < 4 ? j * 32 : hoff + (j - 4) * 32) + (hi ? 16 : 0);
      const v8h* ap = (const v8h*)&At[m16][kb];
      const v8h alo = ap[0];
      const v8h ahi = ap[1];  // +8 halves
      v16h a;
      #pragma unroll
      for (int i = 0; i < 8; ++i) { a[i] = alo[i]; a[8 + i] = ahi[i]; }

      a1 = __builtin_amdgcn_wmma_f32_16x16x32_f16(false, a, false, B1[j],
                                                  (short)0, a1, false, false);
      if (j < 4)
        a2 = __builtin_amdgcn_wmma_f32_16x16x32_f16(false, a, false, B2[j],
                                                    (short)0, a2, false, false);
      else
        a3 = __builtin_amdgcn_wmma_f32_16x16x32_f16(false, a, false, B3[j - 4],
                                                    (short)0, a3, false, false);
    }

    // gates for this lane's 8 rows: contiguous in gb -> two ds_load_b128
    const int sl = s & 127;
    const v4f gA = *(const v4f*)&gb[(sl << 4) + mb];
    const v4f gB = *(const v4f*)&gb[(sl << 4) + mb + 4];

    // epilogue: r = sigmoid(a1); h_t = tanh(a2 + r*a3); h = g*h_t + (1-g)*h
    #pragma unroll
    for (int r = 0; r < 8; ++r) {
      const int m  = mb + r;
      const float gv = (r < 4) ? gA[r] : gB[r - 4];
      const float rg = fast_sigmoid(a1[r]);
      const float ht = fast_tanh(a2[r] + rg * a3[r]);
      const float hn = gv * ht + (1.f - gv) * hreg[r];
      hreg[r] = hn;
      At[m][hstore + n] = (_Float16)hn;  // ping-pong: no WAR with this step
    }
    __syncthreads();  // all reads of this step done before next c overwrite
  }

  // final hidden -> output[b, 0, dir*100 + n]
  if (nok) {
    #pragma unroll
    for (int r = 0; r < 8; ++r) {
      const int m = mb + r;
      out[(size_t)(b0 + m) * 200 + dir * Hq + n] = hreg[r];
    }
  }
}

// ---------------------------------------------------------------------------
extern "C" void kernel_launch(void* const* d_in, const int* in_sizes, int n_in,
                              void* d_out, int out_size, void* d_ws, size_t ws_size,
                              hipStream_t stream) {
  (void)in_sizes; (void)n_in; (void)d_ws; (void)ws_size; (void)out_size;
  const float* query   = (const float*)d_in[0];
  const float* context = (const float*)d_in[1];
  const float* init_h  = (const float*)d_in[2];
  const float* Wr_f  = (const float*)d_in[3];
  const float* br_f  = (const float*)d_in[4];
  const float* Ur_f  = (const float*)d_in[5];
  const float* bur_f = (const float*)d_in[6];
  const float* W_f   = (const float*)d_in[7];
  const float* bw_f  = (const float*)d_in[8];
  const float* U_f   = (const float*)d_in[9];
  const float* bu_f  = (const float*)d_in[10];
  const float* Wr_b  = (const float*)d_in[11];
  const float* br_b  = (const float*)d_in[12];
  const float* Ur_b  = (const float*)d_in[13];
  const float* bur_b = (const float*)d_in[14];
  const float* W_b   = (const float*)d_in[15];
  const float* bw_b  = (const float*)d_in[16];
  const float* U_b   = (const float*)d_in[17];
  const float* bu_b  = (const float*)d_in[18];

  float* out    = (float*)d_out;           // (512,1,200)
  float* scores = out + (size_t)Bq * 200;  // (512,1,1024)

  attn_softmax_kernel<<<Bq, 256, 0, stream>>>(query, context, scores);
  attgru_scan_kernel<<<64, 224, 0, stream>>>(
      context, init_h, scores,
      Wr_f, br_f, Ur_f, bur_f, W_f, bw_f, U_f, bu_f,
      Wr_b, br_b, Ur_b, bur_b, W_b, bw_b, U_b, bu_b,
      out);
}
